// ReformerBlock_82746839924980
// MI455X (gfx1250) — matrix operand order repaired and checked by reading, not compiled, not run on hardware
//
#include <hip/hip_runtime.h>

// ---------------------------------------------------------------------------
// Reformer block for MI455X (gfx1250, wave32, WMMA).
// All GEMMs + both attention matmuls use v_wmma_f32_16x16x32_f16.
// GEMM uses 2x2 register blocking per wave (4 accumulators) to double
// FLOP/byte and expose 4 independent back-to-back WMMAs per K-step.
// Weights are cast f32->f16 and transposed once per call so every WMMA
// B-fragment is a contiguous 32-byte load.  Activations travel as f16
// between GEMMs (f32 accumulate), residual stream stays f32.
// ---------------------------------------------------------------------------

typedef __attribute__((ext_vector_type(16))) _Float16 v16h;
typedef __attribute__((ext_vector_type(8)))  _Float16 v8h;
typedef __attribute__((ext_vector_type(8)))  float    v8f;

union V16 { v16h v; v8h h[2]; };

#define BT      32768     // B*T tokens
#define TSEQ    8192
#define CH      1024
#define NHEAD   16
#define HD      64
#define BUCKET  64
#define NB      128       // buckets per sequence
#define QKVC    3072

// ---------------------------------------------------------------------------
// Weight cast + transpose:  w[K,N] f32  ->  wt[N,K] f16
// ---------------------------------------------------------------------------
__global__ __launch_bounds__(256) void cast_transpose(
    const float* __restrict__ w, _Float16* __restrict__ wt, int K, int N) {
  size_t idx = (size_t)blockIdx.x * 256 + threadIdx.x;
  if (idx >= (size_t)K * N) return;
  int n = (int)(idx / K);
  int k = (int)(idx % K);
  wt[idx] = (_Float16)w[(size_t)k * N + n];
}

// ---------------------------------------------------------------------------
// LayerNorm (biased variance, eps=1e-5), f32 in -> f16 out. One row per block.
// ---------------------------------------------------------------------------
__global__ __launch_bounds__(256) void ln_f16(
    const float* __restrict__ x, const float* __restrict__ g,
    const float* __restrict__ b, _Float16* __restrict__ out) {
  const int row = blockIdx.x;
  const int tid = threadIdx.x;
  const float* xr = x + (size_t)row * CH;
  float v[4];
  float s = 0.f, s2 = 0.f;
#pragma unroll
  for (int i = 0; i < 4; ++i) {
    v[i] = xr[tid + i * 256];
    s += v[i];
    s2 += v[i] * v[i];
  }
  __shared__ float r1[256], r2[256];
  r1[tid] = s; r2[tid] = s2;
  __syncthreads();
  for (int off = 128; off > 0; off >>= 1) {
    if (tid < off) { r1[tid] += r1[tid + off]; r2[tid] += r2[tid + off]; }
    __syncthreads();
  }
  __shared__ float mi[2];
  if (tid == 0) {
    float m = r1[0] * (1.0f / CH);
    float var = r2[0] * (1.0f / CH) - m * m;
    mi[0] = m;
    mi[1] = rsqrtf(var + 1e-5f);
  }
  __syncthreads();
  const float m = mi[0], inv = mi[1];
#pragma unroll
  for (int i = 0; i < 4; ++i) {
    int c = tid + i * 256;
    out[(size_t)row * CH + c] = (_Float16)((v[i] - m) * inv * g[c] + b[c]);
  }
}

// ---------------------------------------------------------------------------
// WMMA GEMM:  out[M,N] = act( A[M,K] * Bt[N,K]^T + bias )  (+ res)
// A, Bt f16 row-major, f32 accumulate.
// Block = 8 waves arranged 2(M) x 4(N); block tile 64(M) x 128(N).
// Each wave computes a 32x32 sub-tile: 2x2 WMMA tiles, 4 accumulators,
// 4 WMMAs per 128 B of fragment loads per lane per K-step.
// ---------------------------------------------------------------------------
template<bool GELU, bool RES, bool OUT16>
__global__ __launch_bounds__(256) void wmma_gemm(
    const _Float16* __restrict__ A, const _Float16* __restrict__ Bt,
    const float* __restrict__ bias, const float* __restrict__ res,
    float* __restrict__ outF, _Float16* __restrict__ outH,
    int M, int N, int K) {
  const int lane = threadIdx.x & 31;
  const int wave = threadIdx.x >> 5;
  const int hi16 = lane >> 4;       // 0: lanes 0-15, 1: lanes 16-31
  const int l15  = lane & 15;
  const int wm   = wave >> 2;       // 0..1
  const int wn   = wave & 3;        // 0..3
  const int m0   = blockIdx.y * 64 + wm * 32;   // tiles m0, m0+16
  const int n0   = blockIdx.x * 128 + wn * 32;  // tiles n0, n0+16

  const _Float16* aptr0 = A  + (size_t)(m0 + l15) * K;
  const _Float16* aptr1 = A  + (size_t)(m0 + 16 + l15) * K;
  const _Float16* bptr0 = Bt + (size_t)(n0 + l15) * K;
  const _Float16* bptr1 = Bt + (size_t)(n0 + 16 + l15) * K;

  v8f c00 = {}, c01 = {}, c10 = {}, c11 = {};
  for (int kk = 0; kk < K; kk += 32) {
    V16 a0, a1, b0, b1;
    const int ka = kk + hi16 * 16;
    const int kb = kk + hi16 * 16;
    a0.h[0] = *(const v8h*)(aptr0 + ka);
    a0.h[1] = *(const v8h*)(aptr0 + ka + 8);
    a1.h[0] = *(const v8h*)(aptr1 + ka);
    a1.h[1] = *(const v8h*)(aptr1 + ka + 8);
    b0.v    = *(const v16h*)(bptr0 + kb);
    b1.v    = *(const v16h*)(bptr1 + kb);
    c00 = __builtin_amdgcn_wmma_f32_16x16x32_f16(
        false, a0.v, false, b0.v, (short)0, c00, false, false);
    c01 = __builtin_amdgcn_wmma_f32_16x16x32_f16(
        false, a0.v, false, b1.v, (short)0, c01, false, false);
    c10 = __builtin_amdgcn_wmma_f32_16x16x32_f16(
        false, a1.v, false, b0.v, (short)0, c10, false, false);
    c11 = __builtin_amdgcn_wmma_f32_16x16x32_f16(
        false, a1.v, false, b1.v, (short)0, c11, false, false);
  }

  const float bias0 = bias[n0 + l15];
  const float bias1 = bias[n0 + 16 + l15];
  const v8f* accs[4] = {&c00, &c01, &c10, &c11};
#pragma unroll
  for (int t = 0; t < 4; ++t) {
    const int mbase = m0 + ((t >> 1) ? 16 : 0);
    const int n     = n0 + ((t & 1) ? 16 : 0) + l15;
    const float bb  = (t & 1) ? bias1 : bias0;
    const v8f&  c   = *accs[t];
#pragma unroll
    for (int r = 0; r < 8; ++r) {
      const int row = mbase + r + hi16 * 8;
      const size_t o = (size_t)row * N + n;
      float val = c[r] + bb;
      if constexpr (RES)  val += res[o];
      if constexpr (GELU) val = 0.5f * val * (1.0f + erff(val * 0.70710678118f));
      if constexpr (OUT16) outH[o] = (_Float16)val;
      else                 outF[o] = val;
    }
  }
}

// ---------------------------------------------------------------------------
// Bucketed local attention, one block per (bucket, head, batch).
//   S = Q Kc^T * scale, mask(prev bucket full | causal current), softmax,
//   O = P Vc.    Q/K/V read from f16 qkv [token][3C] layout.
// ---------------------------------------------------------------------------
__global__ __launch_bounds__(256) void attn_kernel(
    const _Float16* __restrict__ qkv, _Float16* __restrict__ ao) {
  __shared__ _Float16 sK [128 * 64];   // keys  [j][d]   (j<64: prev bucket)
  __shared__ _Float16 sVT[64 * 128];   // V^T   [d][j]
  __shared__ float    sS [64 * 128];   // scores[i][j]
  __shared__ _Float16 sP [64 * 128];   // probs [i][j]

  const int tid    = threadIdx.x;
  const int bucket = blockIdx.x;
  const int head   = blockIdx.y;
  const int bz     = blockIdx.z;
  const size_t base = (size_t)bz * TSEQ * QKVC;
  const int hoff = head * HD;

  // ---- stage K into LDS (8 halves per iteration) -------------------------
  for (int idx = tid; idx < 1024; idx += 256) {
    const int j  = idx >> 3;
    const int d0 = (idx & 7) << 3;
    v8h val = {};
    if (j >= BUCKET || bucket > 0) {
      const int tok = (j < BUCKET) ? ((bucket - 1) * BUCKET + j)
                                   : (bucket * BUCKET + (j - BUCKET));
      val = *(const v8h*)(qkv + base + (size_t)tok * QKVC + CH + hoff + d0);
    }
    *(v8h*)(sK + j * 64 + d0) = val;
  }
  // ---- stage V transposed into LDS ---------------------------------------
  for (int idx = tid; idx < 8192; idx += 256) {
    const int j = idx >> 6;
    const int d = idx & 63;
    _Float16 val = (_Float16)0.f;
    if (j >= BUCKET || bucket > 0) {
      const int tok = (j < BUCKET) ? ((bucket - 1) * BUCKET + j)
                                   : (bucket * BUCKET + (j - BUCKET));
      val = qkv[base + (size_t)tok * QKVC + 2 * CH + hoff + d];
    }
    sVT[d * 128 + j] = val;
  }
  __syncthreads();

  const int lane = tid & 31;
  const int wave = tid >> 5;
  const int hi16 = lane >> 4;
  const int l15  = lane & 15;

  // ---- S = Q K^T : wave w owns key block j0=w*16, loops 4 query tiles ----
  const int j0 = wave * 16;
  for (int mt = 0; mt < 4; ++mt) {
    v8f acc = {};
    const int qrow = mt * 16 + l15;
    const _Float16* qp =
        qkv + base + (size_t)(bucket * BUCKET + qrow) * QKVC + hoff;
    for (int kk = 0; kk < HD; kk += 32) {
      V16 ua, ub;
      const int ka = kk + hi16 * 16;
      ua.h[0] = *(const v8h*)(qp + ka);
      ua.h[1] = *(const v8h*)(qp + ka + 8);
      ub.v    = *(const v16h*)(sK + (j0 + l15) * 64 + kk + hi16 * 16);
      acc = __builtin_amdgcn_wmma_f32_16x16x32_f16(
          false, ua.v, false, ub.v, (short)0, acc, false, false);
    }
    const int j = j0 + l15;
#pragma unroll
    for (int r = 0; r < 8; ++r) {
      const int i = mt * 16 + r + hi16 * 8;       // query index
      const bool vis = (j < BUCKET) || ((j - BUCKET) <= i);
      sS[i * 128 + j] = vis ? acc[r] * 0.125f : -1e30f;
    }
  }
  __syncthreads();

  // ---- softmax over each row of 128 --------------------------------------
  if (tid < 64) {
    float mx = -1e30f;
    for (int j = 0; j < 128; ++j) mx = fmaxf(mx, sS[tid * 128 + j]);
    float sum = 0.f;
    for (int j = 0; j < 128; ++j) {
      const float e = expf(sS[tid * 128 + j] - mx);
      sum += e;
      sP[tid * 128 + j] = (_Float16)e;
    }
    const float inv = 1.0f / sum;
    for (int j = 0; j < 128; ++j)
      sP[tid * 128 + j] = (_Float16)((float)sP[tid * 128 + j] * inv);
  }
  __syncthreads();

  // ---- O = P V : each wave does 2 of the 16 (4x4) output tiles -----------
  const int mt = wave >> 1;
  for (int nt = (wave & 1) * 2; nt <= (wave & 1) * 2 + 1; ++nt) {
    v8f acc = {};
    for (int kk = 0; kk < 128; kk += 32) {
      V16 ua, ub;
      const int ja = kk + hi16 * 16;
      const _Float16* pp = sP + (mt * 16 + l15) * 128 + ja;
      ua.h[0] = *(const v8h*)(pp);
      ua.h[1] = *(const v8h*)(pp + 8);
      ub.v    = *(const v16h*)(sVT + (nt * 16 + l15) * 128 + kk + hi16 * 16);
      acc = __builtin_amdgcn_wmma_f32_16x16x32_f16(
          false, ua.v, false, ub.v, (short)0, acc, false, false);
    }
    const int d = nt * 16 + l15;
#pragma unroll
    for (int r = 0; r < 8; ++r) {
      const int i   = mt * 16 + r + hi16 * 8;
      const int tok = bucket * BUCKET + i;
      ao[((size_t)bz * TSEQ + tok) * CH + hoff + d] = (_Float16)acc[r];
    }
  }
}

// ---------------------------------------------------------------------------
extern "C" void kernel_launch(void* const* d_in, const int* in_sizes, int n_in,
                              void* d_out, int out_size, void* d_ws, size_t ws_size,
                              hipStream_t stream) {
  const float* x     = (const float*)d_in[0];
  const float* ln1_g = (const float*)d_in[1];
  const float* ln1_b = (const float*)d_in[2];
  const float* w_qkv = (const float*)d_in[3];
  const float* b_qkv = (const float*)d_in[4];
  const float* w_out = (const float*)d_in[5];
  const float* b_out = (const float*)d_in[6];
  const float* ln2_g = (const float*)d_in[7];
  const float* ln2_b = (const float*)d_in[8];
  const float* w1    = (const float*)d_in[9];
  const float* b1    = (const float*)d_in[10];
  const float* w2    = (const float*)d_in[11];
  const float* b2    = (const float*)d_in[12];

  // workspace bump allocator (256B aligned)
  char* ws = (char*)d_ws;
  size_t off = 0;
  auto alloc = [&](size_t bytes) -> void* {
    void* p = ws + off;
    off += (bytes + 255) & ~(size_t)255;
    return p;
  };
  _Float16* wqkvT = (_Float16*)alloc((size_t)QKVC * CH * 2);      // [3072,1024]
  _Float16* woutT = (_Float16*)alloc((size_t)CH * CH * 2);        // [1024,1024]
  _Float16* w1T   = (_Float16*)alloc((size_t)4096 * CH * 2);      // [4096,1024]
  _Float16* w2T   = (_Float16*)alloc((size_t)CH * 4096 * 2);      // [1024,4096]
  _Float16* hbuf  = (_Float16*)alloc((size_t)BT * CH * 2);        // LN1 / LN2 out
  _Float16* qkv   = (_Float16*)alloc((size_t)BT * 4096 * 2);      // qkv, reused as m1
  _Float16* ao    = (_Float16*)alloc((size_t)BT * CH * 2);        // attention out
  float*    x2    = (float*)alloc((size_t)BT * CH * 4);           // residual stream
  _Float16* m1    = qkv;                                          // MLP hidden (f16)

  // 1) cast+transpose all weights to f16
  cast_transpose<<<(CH * QKVC) / 256, 256, 0, stream>>>(w_qkv, wqkvT, CH, QKVC);
  cast_transpose<<<(CH * CH) / 256,   256, 0, stream>>>(w_out, woutT, CH, CH);
  cast_transpose<<<(CH * 4096) / 256, 256, 0, stream>>>(w1, w1T, CH, 4096);
  cast_transpose<<<(4096 * CH) / 256, 256, 0, stream>>>(w2, w2T, 4096, CH);

  // 2) h1 = LN1(x) as f16
  ln_f16<<<BT, 256, 0, stream>>>(x, ln1_g, ln1_b, hbuf);

  // 3) qkv = h1 @ w_qkv + b_qkv   (f16 out)
  wmma_gemm<false, false, true><<<dim3(QKVC / 128, BT / 64), 256, 0, stream>>>(
      hbuf, wqkvT, b_qkv, nullptr, nullptr, qkv, BT, QKVC, CH);

  // 4) bucketed local attention -> ao (f16)
  attn_kernel<<<dim3(NB, NHEAD, 4), 256, 0, stream>>>(qkv, ao);

  // 5) x2 = x + ao @ w_out + b_out   (f32)
  wmma_gemm<false, true, false><<<dim3(CH / 128, BT / 64), 256, 0, stream>>>(
      ao, woutT, b_out, x, x2, nullptr, BT, CH, CH);

  // 6) h2 = LN2(x2) as f16
  ln_f16<<<BT, 256, 0, stream>>>(x2, ln2_g, ln2_b, hbuf);

  // 7) m1 = gelu(h2 @ w1 + b1)   (f16)
  wmma_gemm<true, false, true><<<dim3(4096 / 128, BT / 64), 256, 0, stream>>>(
      hbuf, w1T, b1, nullptr, nullptr, m1, BT, 4096, CH);

  // 8) out = x2 + m1 @ w2 + b2   (f32)
  wmma_gemm<false, true, false><<<dim3(CH / 128, BT / 64), 256, 0, stream>>>(
      m1, w2T, b2, x2, (float*)d_out, nullptr, BT, CH, 4096);
}
